// WindowAttention_89593017795200
// MI455X (gfx1250) — compile-verified, matrix operand order checked by reading, not run on hardware
//
#include <hip/hip_runtime.h>
#include <hip/hip_bf16.h>

typedef _Float16 half_t;
typedef __attribute__((ext_vector_type(16))) _Float16 v16h;
typedef __attribute__((ext_vector_type(8)))  float    v8f;

#define WMMA_F16(a, b, c) \
  __builtin_amdgcn_wmma_f32_16x16x32_f16(false, (a), false, (b), (short)0, (c), false, false)

constexpr int BATCH = 2048;
constexpr int NPIX  = 64;     // 8x8 window
constexpr int DIMC  = 192;
constexpr int HEADS = 6;
constexpr int CHEAD = 32;
constexpr int QKVC  = 576;    // 3*DIMC
constexpr int NWIN  = 512;

// GEMM wave blocking: 2 M-tiles x 4 N-tiles of 16x16 per wave.
constexpr int MT = 2;
constexpr int NT = 4;

// ---------------------------------------------------------------------------
// Weight prep: fp32 -> f16, row-major [oc][ic] (B-fragment friendly).
// ---------------------------------------------------------------------------
__global__ void prep_weights(const float* __restrict__ qkv_w,
                             const float* __restrict__ proj_w,
                             half_t* __restrict__ wqkv,
                             half_t* __restrict__ wproj) {
  int i = blockIdx.x * blockDim.x + threadIdx.x;
  if (i < QKVC * DIMC) wqkv[i]  = (half_t)qkv_w[i];
  if (i < DIMC * DIMC) wproj[i] = (half_t)proj_w[i];
}

// ---------------------------------------------------------------------------
// QKV 1x1 conv as WMMA GEMM: Y[M=131072][N=576] = X[M][192] * W^T + bias.
// Wave computes a 32x64 output block: K=192 -> 6 chunks x 8 WMMAs.
// ---------------------------------------------------------------------------
__global__ __launch_bounds__(32)
void gemm_qkv(const float* __restrict__ A, const half_t* __restrict__ W,
              const float* __restrict__ bias, half_t* __restrict__ Y) {
  const int lane = threadIdx.x & 31;
  const int lo = lane & 15;
  const int kh = (lane >> 4) << 3;            // 0 or 8 (K sub-block per half-wave)
  const int n0 = blockIdx.x * 16 * NT;
  const int m0 = blockIdx.y * 16 * MT;

  const float* arow[MT];
#pragma unroll
  for (int mt = 0; mt < MT; ++mt) arow[mt] = A + (size_t)(m0 + 16 * mt + lo) * DIMC;
  const half_t* brow[NT];
#pragma unroll
  for (int t = 0; t < NT; ++t) brow[t] = W + (size_t)(n0 + 16 * t + lo) * DIMC;

  v8f acc[MT][NT];
#pragma unroll
  for (int mt = 0; mt < MT; ++mt)
#pragma unroll
    for (int t = 0; t < NT; ++t) acc[mt][t] = v8f{};

#pragma unroll
  for (int k0 = 0; k0 < DIMC; k0 += 32) {
    v16h a[MT];
#pragma unroll
    for (int mt = 0; mt < MT; ++mt) {
      const float* ap = arow[mt] + k0 + kh;
      v16h av = {};
#pragma unroll
      for (int i = 0; i < 8; ++i) { av[i] = (half_t)ap[i]; av[i + 8] = (half_t)ap[i + 16]; }
      a[mt] = av;
    }
#pragma unroll
    for (int t = 0; t < NT; ++t) {
      const half_t* bp = brow[t] + k0 + kh;
      v16h b = {};
#pragma unroll
      for (int i = 0; i < 8; ++i) { b[i] = bp[i]; b[i + 8] = bp[i + 16]; }
#pragma unroll
      for (int mt = 0; mt < MT; ++mt) acc[mt][t] = WMMA_F16(a[mt], b, acc[mt][t]);
    }
  }

#pragma unroll
  for (int t = 0; t < NT; ++t) {
    const int col = n0 + 16 * t + lo;
    const float bv = bias[col];
#pragma unroll
    for (int mt = 0; mt < MT; ++mt)
#pragma unroll
      for (int j = 0; j < 8; ++j) {
        int row = m0 + 16 * mt + ((lane >> 4) << 3) + j;   // D: 8 rows per lane-half
        Y[(size_t)row * QKVC + col] = (half_t)(acc[mt][t][j] + bv);
      }
  }
}

// ---------------------------------------------------------------------------
// Depthwise 3x3 conv + bias, L2-normalize q/k per (head, pixel) via wave32
// shuffle reduction (lanes == the 32 channels of one head), pack:
//   Qp, Kp : [b][head][n=64][c=32]  (A/B fragment contiguous)
//   Vt     : [b][head][c=32][m=64]  (B fragment for P·V contiguous)
// ---------------------------------------------------------------------------
__global__ __launch_bounds__(256)
void dw_norm_pack(const half_t* __restrict__ qkv, const float* __restrict__ dw_w,
                  const float* __restrict__ dw_b,
                  half_t* __restrict__ Qp, half_t* __restrict__ Kp,
                  half_t* __restrict__ Vt) {
  const int b    = blockIdx.x;
  const int lane = threadIdx.x & 31;   // channel within 32-ch group
  const int py   = threadIdx.x >> 5;   // pixel row, constant per wave
  const half_t* base = qkv + (size_t)b * NPIX * QKVC;

  for (int cb = 0; cb < 18; ++cb) {    // 18 groups of 32 channels (q:0-5 k:6-11 v:12-17)
    const int oc = cb * 32 + lane;
    float w9[9];
#pragma unroll
    for (int t = 0; t < 9; ++t) w9[t] = dw_w[oc * 9 + t];
    const float bv = dw_b[oc];
#pragma unroll
    for (int px = 0; px < 8; ++px) {
      float acc = bv;
#pragma unroll
      for (int ky = 0; ky < 3; ++ky) {
        int y = py + ky - 1;
        if ((unsigned)y < 8u) {
#pragma unroll
          for (int kx = 0; kx < 3; ++kx) {
            int xx = px + kx - 1;
            if ((unsigned)xx < 8u)
              acc += (float)base[(size_t)(y * 8 + xx) * QKVC + oc] * w9[ky * 3 + kx];
          }
        }
      }
      const int p = py * 8 + px;
      if (cb < 12) {                    // q or k -> cosine normalize over 32 lanes
        float ss = acc * acc;
#pragma unroll
        for (int off = 16; off; off >>= 1) ss += __shfl_xor(ss, off, 32);
        float vn = acc / fmaxf(sqrtf(ss), 1e-12f);
        if (cb < 6)
          Qp[(((size_t)b * HEADS + cb)       * NPIX + p) * CHEAD + lane] = (half_t)vn;
        else
          Kp[(((size_t)b * HEADS + (cb - 6)) * NPIX + p) * CHEAD + lane] = (half_t)vn;
      } else {
        Vt[(((size_t)b * HEADS + (cb - 12)) * CHEAD + lane) * NPIX + p] = (half_t)acc;
      }
    }
  }
}

// ---------------------------------------------------------------------------
// Attention per (b, head). 4 waves; wave s owns rows [16s, 16s+16).
//  S = q·k^T : 4 WMMAs (K=32 exactly). Fused temp * S + rel_bias + mask,
//  register softmax (shfl over the 16-lane row group), P -> LDS -> A-frags,
//  O = P·V : 4 WMMAs (2 c-tiles x 2 K-chunks). Output [b][n][head*32+c] f16.
// ---------------------------------------------------------------------------
__global__ __launch_bounds__(128)
void attention(const half_t* __restrict__ Qp, const half_t* __restrict__ Kp,
               const half_t* __restrict__ Vt, const float* __restrict__ mask,
               const float* __restrict__ temperature,
               const float* __restrict__ rel_table,
               half_t* __restrict__ Yo) {
  const int bh = blockIdx.x;
  const int b = bh / HEADS, h = bh - b * HEADS;
  const int lane = threadIdx.x & 31;
  const int s  = threadIdx.x >> 5;  // strip
  const int hi = lane >> 4;
  const int lo = lane & 15;
  const int kh = hi << 3;

  __shared__ half_t Pl[4][16 * 64];

  const half_t* qb = Qp + (size_t)bh * NPIX * CHEAD;
  const half_t* kb = Kp + (size_t)bh * NPIX * CHEAD;
  const half_t* vb = Vt + (size_t)bh * CHEAD * NPIX;

  // Q A-fragment for this strip (rows 16s..16s+15, K=32)
  v16h aq = {};
  {
    const half_t* qp = qb + (size_t)(16 * s + lo) * CHEAD + kh;
#pragma unroll
    for (int i = 0; i < 8; ++i) { aq[i] = qp[i]; aq[i + 8] = qp[i + 16]; }
  }

  // S strip: 4 tiles of 16x16, one WMMA each
  v8f st[4];
#pragma unroll
  for (int t = 0; t < 4; ++t) {
    v16h bk = {};
    const half_t* kp = kb + (size_t)(16 * t + lo) * CHEAD + kh;
#pragma unroll
    for (int i = 0; i < 8; ++i) { bk[i] = kp[i]; bk[i + 8] = kp[i + 16]; }
    v8f z = {};
    st[t] = WMMA_F16(aq, bk, z);
  }

  const float temp = temperature[h];
  const float* mrow = mask + (size_t)(b & (NWIN - 1)) * NPIX * NPIX;

  float pv[4][8];
#pragma unroll
  for (int j = 0; j < 8; ++j) {
    const int n = 16 * s + 8 * hi + j;
    const int ny = n >> 3, nx = n & 7;
    float mx = -3.0e38f;
#pragma unroll
    for (int t = 0; t < 4; ++t) {
      const int m = 16 * t + lo;
      const int ridx = (ny - (m >> 3) + 7) * 15 + (nx - (m & 7) + 7);
      float v = st[t][j] * temp + rel_table[ridx * HEADS + h] + mrow[n * NPIX + m];
      pv[t][j] = v;
      mx = fmaxf(mx, v);
    }
#pragma unroll
    for (int off = 8; off; off >>= 1) mx = fmaxf(mx, __shfl_xor(mx, off, 32));
    float sum = 0.f;
#pragma unroll
    for (int t = 0; t < 4; ++t) { float e = __expf(pv[t][j] - mx); pv[t][j] = e; sum += e; }
#pragma unroll
    for (int off = 8; off; off >>= 1) sum += __shfl_xor(sum, off, 32);
    const float inv = 1.f / sum;
#pragma unroll
    for (int t = 0; t < 4; ++t) pv[t][j] *= inv;
  }

  // D-layout -> A-fragment reshape via LDS
#pragma unroll
  for (int t = 0; t < 4; ++t)
#pragma unroll
    for (int j = 0; j < 8; ++j)
      Pl[s][(8 * hi + j) * 64 + 16 * t + lo] = (half_t)pv[t][j];
  __syncthreads();

  v8f o[2] = {v8f{}, v8f{}};
#pragma unroll
  for (int kc = 0; kc < 2; ++kc) {
    v16h ap = {};
    const half_t* pp = &Pl[s][lo * 64 + kc * 32 + kh];
#pragma unroll
    for (int i = 0; i < 8; ++i) { ap[i] = pp[i]; ap[i + 8] = pp[i + 16]; }
#pragma unroll
    for (int t = 0; t < 2; ++t) {
      v16h bvf = {};
      const half_t* vp = vb + (size_t)(16 * t + lo) * NPIX + kc * 32 + kh;
#pragma unroll
      for (int i = 0; i < 8; ++i) { bvf[i] = vp[i]; bvf[i + 8] = vp[i + 16]; }
      o[t] = WMMA_F16(ap, bvf, o[t]);
    }
  }

#pragma unroll
  for (int t = 0; t < 2; ++t)
#pragma unroll
    for (int j = 0; j < 8; ++j) {
      const int n = 16 * s + 8 * hi + j;
      const int c = 16 * t + lo;
      Yo[((size_t)b * NPIX + n) * DIMC + h * CHEAD + c] = (half_t)o[t][j];
    }
}

// ---------------------------------------------------------------------------
// Proj 1x1 conv: out[b][oc][p] = sum_ic Yo[b][p][ic] * Wp[oc][ic] + bias.
// Wave computes 32x64 output block; fp32 NCHW-transposed store into d_out.
// ---------------------------------------------------------------------------
__global__ __launch_bounds__(32)
void gemm_proj(const half_t* __restrict__ A, const half_t* __restrict__ W,
               const float* __restrict__ bias, float* __restrict__ out) {
  const int lane = threadIdx.x & 31;
  const int lo = lane & 15;
  const int kh = (lane >> 4) << 3;
  const int n0 = blockIdx.x * 16 * NT;
  const int m0 = blockIdx.y * 16 * MT;

  const half_t* arow[MT];
#pragma unroll
  for (int mt = 0; mt < MT; ++mt) arow[mt] = A + (size_t)(m0 + 16 * mt + lo) * DIMC;
  const half_t* brow[NT];
#pragma unroll
  for (int t = 0; t < NT; ++t) brow[t] = W + (size_t)(n0 + 16 * t + lo) * DIMC;

  v8f acc[MT][NT];
#pragma unroll
  for (int mt = 0; mt < MT; ++mt)
#pragma unroll
    for (int t = 0; t < NT; ++t) acc[mt][t] = v8f{};

#pragma unroll
  for (int k0 = 0; k0 < DIMC; k0 += 32) {
    v16h a[MT];
#pragma unroll
    for (int mt = 0; mt < MT; ++mt) {
      const half_t* ap = arow[mt] + k0 + kh;
      v16h av = {};
#pragma unroll
      for (int i = 0; i < 8; ++i) { av[i] = ap[i]; av[i + 8] = ap[i + 16]; }
      a[mt] = av;
    }
#pragma unroll
    for (int t = 0; t < NT; ++t) {
      const half_t* bp = brow[t] + k0 + kh;
      v16h b = {};
#pragma unroll
      for (int i = 0; i < 8; ++i) { b[i] = bp[i]; b[i + 8] = bp[i + 16]; }
#pragma unroll
      for (int mt = 0; mt < MT; ++mt) acc[mt][t] = WMMA_F16(a[mt], b, acc[mt][t]);
    }
  }

#pragma unroll
  for (int t = 0; t < NT; ++t) {
    const int col = n0 + 16 * t + lo;
    const float bv = bias[col];
#pragma unroll
    for (int mt = 0; mt < MT; ++mt)
#pragma unroll
      for (int j = 0; j < 8; ++j) {
        const int row = m0 + 16 * mt + ((lane >> 4) << 3) + j;
        const int bb = row >> 6, p = row & 63;
        out[((size_t)bb * DIMC + col) * NPIX + p] = acc[mt][t][j] + bv;
      }
  }
}

// ---------------------------------------------------------------------------
extern "C" void kernel_launch(void* const* d_in, const int* in_sizes, int n_in,
                              void* d_out, int out_size, void* d_ws, size_t ws_size,
                              hipStream_t stream) {
  (void)in_sizes; (void)n_in; (void)out_size; (void)ws_size;
  const float* x           = (const float*)d_in[0];
  const float* mask        = (const float*)d_in[1];
  const float* qkv_w       = (const float*)d_in[2];
  const float* qkv_b       = (const float*)d_in[3];
  const float* dw_w        = (const float*)d_in[4];
  const float* dw_b        = (const float*)d_in[5];
  const float* proj_w      = (const float*)d_in[6];
  const float* proj_b      = (const float*)d_in[7];
  const float* temperature = (const float*)d_in[8];
  const float* rel_table   = (const float*)d_in[9];
  float* out = (float*)d_out;

  char* ws = (char*)d_ws;
  size_t off = 0;
  auto carve = [&](size_t bytes) -> char* {
    char* p = ws + off;
    off = (off + bytes + 255) & ~(size_t)255;
    return p;
  };
  half_t* Wqkv  = (half_t*)carve((size_t)QKVC * DIMC * sizeof(half_t));
  half_t* Wproj = (half_t*)carve((size_t)DIMC * DIMC * sizeof(half_t));
  half_t* qkv_h = (half_t*)carve((size_t)BATCH * NPIX * QKVC * sizeof(half_t));
  half_t* Qp    = (half_t*)carve((size_t)BATCH * HEADS * NPIX * CHEAD * sizeof(half_t));
  half_t* Kp    = (half_t*)carve((size_t)BATCH * HEADS * NPIX * CHEAD * sizeof(half_t));
  half_t* Vt    = (half_t*)carve((size_t)BATCH * HEADS * NPIX * CHEAD * sizeof(half_t));
  half_t* Yo    = qkv_h;  // reuse: qkv_h is dead after dw_norm_pack

  prep_weights<<<(QKVC * DIMC + 255) / 256, 256, 0, stream>>>(qkv_w, proj_w, Wqkv, Wproj);

  dim3 gA(QKVC / (16 * NT), BATCH * NPIX / (16 * MT));   // (9, 4096)
  gemm_qkv<<<gA, 32, 0, stream>>>(x, Wqkv, qkv_b, qkv_h);

  dw_norm_pack<<<BATCH, 256, 0, stream>>>(qkv_h, dw_w, dw_b, Qp, Kp, Vt);

  attention<<<BATCH * HEADS, 128, 0, stream>>>(Qp, Kp, Vt, mask, temperature, rel_table, Yo);

  dim3 gD(DIMC / (16 * NT), BATCH * NPIX / (16 * MT));   // (3, 4096)
  gemm_proj<<<gD, 32, 0, stream>>>(Yo, Wproj, proj_b, out);
}
